// Head_10471130268040
// MI455X (gfx1250) — matrix operand order repaired and checked by reading, not compiled, not run on hardware
//
#include <hip/hip_runtime.h>
#include <hip/hip_bf16.h>

typedef __bf16 bf16;
typedef __attribute__((ext_vector_type(16))) __bf16 v16bf;
typedef __attribute__((ext_vector_type(8)))  __bf16 v8bf;
typedef __attribute__((ext_vector_type(8)))  float   v8f;
typedef __attribute__((ext_vector_type(4)))  float   v4f;

#define BATCH 256
#define SEQ   320
#define EMBD  1024
#define HEAD  64

__device__ __forceinline__ v16bf load16bf(const bf16* p) {
  v8bf a = *(const v8bf*)p;
  v8bf b = *(const v8bf*)(p + 8);
  v16bf r;
#pragma unroll
  for (int i = 0; i < 8; ++i) { r[i] = a[i]; r[i + 8] = b[i]; }
  return r;
}

__device__ __forceinline__ v8f wmma_bf16(v16bf a, v16bf b, v8f c) {
  // D = A(16x32 bf16) x B(32x16 bf16) + C(16x16 f32)
  return __builtin_amdgcn_wmma_f32_16x16x32_bf16(false, a, false, b,
                                                 (short)0, c, false, false);
}

// ---------------------------------------------------------------------------
// Kernel 0: transpose + convert weights to bf16: Wt[w][h][c], w: 0=Q,1=K,2=V
// ---------------------------------------------------------------------------
__global__ void wt_convert_kernel(const float* __restrict__ Wq,
                                  const float* __restrict__ Wk,
                                  const float* __restrict__ Wv,
                                  bf16* __restrict__ Wt) {
  int idx = blockIdx.x * blockDim.x + threadIdx.x;   // 3*64*1024 threads
  if (idx >= 3 * HEAD * EMBD) return;
  int w   = idx >> 16;          // / (64*1024)
  int rem = idx & 0xFFFF;
  int h   = rem >> 10;          // / 1024
  int c   = rem & 1023;
  const float* W = (w == 0) ? Wq : (w == 1) ? Wk : Wv;
  Wt[idx] = (bf16)W[c * HEAD + h];
}

// ---------------------------------------------------------------------------
// Kernel 1: fused QKV projection, register-blocked 2x on M.
// Each wave owns 32 rows of x (two 16-row A tiles); every weight B tile
// loaded once per K-step feeds two WMMAs.  Outputs: q,k as bf16 [b*T+t][h]
// (q pre-scaled by C^-0.5 = 1/32), v transposed vT[b][h][t] bf16.
// ---------------------------------------------------------------------------
__global__ void __launch_bounds__(128)
qkv_kernel(const float* __restrict__ x, const bf16* __restrict__ Wt,
           bf16* __restrict__ q, bf16* __restrict__ k, bf16* __restrict__ vT) {
  const int lane = threadIdx.x & 31;
  const int wave = threadIdx.x >> 5;
  const int mt2  = blockIdx.x * 4 + wave;      // 0..2559, 32 rows each
  const int m    = lane & 15;
  const int hi   = lane >> 4;

  v8f acc[24];                                  // [a(2)][nt(12)]
#pragma unroll
  for (int i = 0; i < 24; ++i) acc[i] = (v8f)0.f;

  const float* xr[2];
  xr[0] = x + (size_t)(mt2 * 32 + m) * EMBD;
  xr[1] = xr[0] + (size_t)16 * EMBD;

  for (int kt = 0; kt < EMBD / 32; ++kt) {
    if (kt + 1 < EMBD / 32) {
      __builtin_prefetch(xr[0] + (kt + 1) * 32 + 8 * hi, 0, 3);
      __builtin_prefetch(xr[1] + (kt + 1) * 32 + 8 * hi, 0, 3);
    }
    // Two A tiles (x rows, fp32 -> bf16).
    // A layout: e=0..15 <- K=16*hi+0..15
    v16bf A[2];
#pragma unroll
    for (int a = 0; a < 2; ++a) {
      const float* p0 = xr[a] + kt * 32 + 16 * hi;
      const float* p1 = p0 + 8;
      v4f a0 = *(const v4f*)p0;
      v4f a1 = *(const v4f*)(p0 + 4);
      v4f a2 = *(const v4f*)p1;
      v4f a3 = *(const v4f*)(p1 + 4);
#pragma unroll
      for (int i = 0; i < 4; ++i) {
        A[a][i]      = (bf16)a0[i];
        A[a][4 + i]  = (bf16)a1[i];
        A[a][8 + i]  = (bf16)a2[i];
        A[a][12 + i] = (bf16)a3[i];
      }
    }
#pragma unroll
    for (int nt = 0; nt < 12; ++nt) {
      // B^T row-major: Wt row (nt*16 + n), 16 contiguous K at 32*kt + 16*hi
      const bf16* bp = Wt + (size_t)(nt * 16 + m) * EMBD + kt * 32 + 16 * hi;
      v16bf Bt = load16bf(bp);
      acc[nt]      = wmma_bf16(A[0], Bt, acc[nt]);
      acc[12 + nt] = wmma_bf16(A[1], Bt, acc[12 + nt]);
    }
  }

  const int batch = (mt2 * 32) / SEQ;          // 32-row group never straddles (32|320)
  const int tbase = mt2 * 32 - batch * SEQ;

#pragma unroll
  for (int a = 0; a < 2; ++a) {
    const int row0  = mt2 * 32 + a * 16;       // global row of this tile
    const int tloc0 = tbase + a * 16;
    // q (scaled) and k: scatter rows, column h = nt*16 + n
#pragma unroll
    for (int nt = 0; nt < 8; ++nt) {
      const int h   = (nt & 3) * 16 + m;
      bf16* dst     = (nt < 4) ? q : k;
      const float s = (nt < 4) ? 0.03125f : 1.0f;   // 1024^-0.5
      const v8f av  = acc[a * 12 + nt];
#pragma unroll
      for (int r = 0; r < 8; ++r) {
        int trow = row0 + r + 8 * hi;
        dst[(size_t)trow * HEAD + h] = (bf16)(av[r] * s);
      }
    }
    // v transposed: vT[b][h][t]; per lane t = tloc0+8*hi+r contiguous -> b128 store
#pragma unroll
    for (int nt = 8; nt < 12; ++nt) {
      const int h  = (nt - 8) * 16 + m;
      const v8f av = acc[a * 12 + nt];
      v8bf pk;
#pragma unroll
      for (int r = 0; r < 8; ++r) pk[r] = (bf16)av[r];
      *(v8bf*)(vT + (size_t)(batch * HEAD + h) * SEQ + tloc0 + 8 * hi) = pk;
    }
  }
}

// ---------------------------------------------------------------------------
// Kernel 2: flash attention.  One wave per (batch, 16-row q tile).
// S = q k^T (q pre-scaled), causal mask, online softmax, O += P v.
// ---------------------------------------------------------------------------
__global__ void __launch_bounds__(128)
attn_kernel(const bf16* __restrict__ q, const bf16* __restrict__ k,
            const bf16* __restrict__ vT, float* __restrict__ out) {
  __shared__ __align__(128) bf16 Plds[4 * 16 * 32];   // per-wave 16x32 P staging

  const int lane = threadIdx.x & 31;
  const int wave = threadIdx.x >> 5;
  const int unit = blockIdx.x * 4 + wave;             // 0..5119
  const int b    = unit / (SEQ / 16);
  const int qt   = unit % (SEQ / 16);
  const int t0   = qt * 16;
  const int m    = lane & 15;
  const int hi   = lane >> 4;

  // Q as two A tiles (K = H = 64)
  const bf16* qrow = q + (size_t)(b * SEQ + t0 + m) * HEAD;
  v16bf Qa[2];
#pragma unroll
  for (int kt = 0; kt < 2; ++kt) {
    v8bf c0 = *(const v8bf*)(qrow + kt * 32 + 16 * hi);
    v8bf c1 = *(const v8bf*)(qrow + kt * 32 + 8 + 16 * hi);
#pragma unroll
    for (int i = 0; i < 8; ++i) { Qa[kt][i] = c0[i]; Qa[kt][i + 8] = c1[i]; }
  }

  float Mx[8], L[8];
  v8f O[4];
#pragma unroll
  for (int r = 0; r < 8; ++r) { Mx[r] = -__builtin_inff(); L[r] = 0.f; }
#pragma unroll
  for (int ht = 0; ht < 4; ++ht) O[ht] = (v8f)0.f;

  bf16* pl = Plds + wave * (16 * 32);
  const int nchunks = (t0 + 15) / 32 + 1;

  for (int j = 0; j < nchunks; ++j) {
    const int s0 = j * 32;
    // ---- S = Q K^T over a 16x32 score block (two 16x16 n-tiles) ----
    v8f S[2];
#pragma unroll
    for (int jj = 0; jj < 2; ++jj) {
      const bf16* kr = k + (size_t)(b * SEQ + s0 + jj * 16 + m) * HEAD + 16 * hi;
      v16bf Kb0 = load16bf(kr);        // h = 16*hi .. +15   (k-tile 0)
      v16bf Kb1 = load16bf(kr + 32);   // h = 32+16*hi ..    (k-tile 1)
      v8f s = (v8f)0.f;
      s = wmma_bf16(Qa[0], Kb0, s);
      s = wmma_bf16(Qa[1], Kb1, s);
      S[jj] = s;
    }
    // ---- causal mask (only on the diagonal chunk) ----
    if (s0 + 31 > t0) {
#pragma unroll
      for (int jj = 0; jj < 2; ++jj) {
        const int sidx = s0 + jj * 16 + m;
#pragma unroll
        for (int r = 0; r < 8; ++r) {
          if (sidx > t0 + r + 8 * hi) S[jj][r] = -__builtin_inff();
        }
      }
    }
    // ---- online softmax: row stats live per-r, reduced across 16 lanes ----
    float alpha[8];
#pragma unroll
    for (int r = 0; r < 8; ++r) {
      float v = fmaxf(S[0][r], S[1][r]);
#pragma unroll
      for (int off = 1; off < 16; off <<= 1) v = fmaxf(v, __shfl_xor(v, off, 32));
      float nm = fmaxf(Mx[r], v);
      alpha[r] = __expf(Mx[r] - nm);
      Mx[r] = nm;
    }
#pragma unroll
    for (int r = 0; r < 8; ++r) {
      float p0 = __expf(S[0][r] - Mx[r]);
      float p1 = __expf(S[1][r] - Mx[r]);
      S[0][r] = p0; S[1][r] = p1;
      float sm = p0 + p1;
#pragma unroll
      for (int off = 1; off < 16; off <<= 1) sm += __shfl_xor(sm, off, 32);
      L[r] = L[r] * alpha[r] + sm;
      O[0][r] *= alpha[r]; O[1][r] *= alpha[r];
      O[2][r] *= alpha[r]; O[3][r] *= alpha[r];
    }
    // ---- P: C-layout -> LDS row-major [m][32] -> reload as A-layout ----
#pragma unroll
    for (int jj = 0; jj < 2; ++jj)
#pragma unroll
      for (int r = 0; r < 8; ++r)
        pl[(r + 8 * hi) * 32 + jj * 16 + m] = (bf16)S[jj][r];
    asm volatile("s_wait_dscnt 0" ::: "memory");   // wave-private LDS, in-order DS
    const bf16* pr = pl + m * 32;
    v16bf Pa;
    {
      v8bf c0 = *(const v8bf*)(pr + 16 * hi);
      v8bf c1 = *(const v8bf*)(pr + 8 + 16 * hi);
#pragma unroll
      for (int i = 0; i < 8; ++i) { Pa[i] = c0[i]; Pa[i + 8] = c1[i]; }
    }
    // ---- O += P V : vT is B^T row-major [h][t] ----
#pragma unroll
    for (int ht = 0; ht < 4; ++ht) {
      const bf16* vr = vT + (size_t)(b * HEAD + ht * 16 + m) * SEQ + s0 + 16 * hi;
      v16bf Vb = load16bf(vr);
      O[ht] = wmma_bf16(Pa, Vb, O[ht]);
    }
  }

  // ---- epilogue: out[b][t][h] = O / L ----
#pragma unroll
  for (int ht = 0; ht < 4; ++ht)
#pragma unroll
    for (int r = 0; r < 8; ++r) {
      int t = t0 + r + 8 * hi;
      out[(size_t)(b * SEQ + t) * HEAD + ht * 16 + m] = O[ht][r] / L[r];
    }
}

// ---------------------------------------------------------------------------
extern "C" void kernel_launch(void* const* d_in, const int* in_sizes, int n_in,
                              void* d_out, int out_size, void* d_ws, size_t ws_size,
                              hipStream_t stream) {
  const float* x  = (const float*)d_in[0];
  const float* Wk = (const float*)d_in[1];
  const float* Wq = (const float*)d_in[2];
  const float* Wv = (const float*)d_in[3];
  float* out = (float*)d_out;

  char* ws = (char*)d_ws;
  bf16* Wt = (bf16*)ws;                                   // 3*64*1024*2 = 384 KB
  bf16* qb = (bf16*)(ws + 3 * HEAD * EMBD * 2);
  bf16* kb = qb + (size_t)BATCH * SEQ * HEAD;             // 10 MB each
  bf16* vb = kb + (size_t)BATCH * SEQ * HEAD;

  wt_convert_kernel<<<(3 * HEAD * EMBD + 255) / 256, 256, 0, stream>>>(Wq, Wk, Wv, Wt);

  const int munits = BATCH * SEQ / 32;                    // 2560 wave units (32 rows)
  qkv_kernel<<<munits / 4, 128, 0, stream>>>(x, Wt, qb, kb, vb);

  const int units = BATCH * (SEQ / 16);                   // 5120 wave units
  attn_kernel<<<units / 4, 128, 0, stream>>>(qb, kb, vb, out);
}
